// MultiBondFastRetention_88132728914057
// MI455X (gfx1250) — matrix operand order repaired and checked by reading, not compiled, not run on hardware
//
#include <hip/hip_runtime.h>
#include <stdint.h>

#define H        256
#define NB       256
#define MMOL     512
#define TOTROWS  (MMOL * NB)          // 131072
#define RPB      32                   // rows per block tile
#define LDSP     264                  // LDS pitch in bf16 elems (256 + 8 pad -> 4-bank row skew)
#define SCALE_F  0.0625f              // 256^-0.5
#define EPS_F    1e-5f
#define LNG_F    (-0.031748698f)      // ln(0.96875)
#define DECAY_F  2.952559e-04f        // 0.96875^256
#define CORR_WIN 3                    // decay^3 ~ 2.6e-11 << f32 eps

typedef __attribute__((ext_vector_type(16))) __bf16 v16bf;
typedef __attribute__((ext_vector_type(8)))  float  v8f;
typedef __attribute__((address_space(3))) unsigned short lds_us;

typedef union { v16bf v; uint4 q[2]; } Frag;

__device__ __forceinline__ unsigned short f2bf(float f) {
    union { float f; uint32_t u; } x; x.f = f;
    uint32_t r = x.u + 0x7FFFu + ((x.u >> 16) & 1u);
    return (unsigned short)(r >> 16);
}

// Preload the 8 K-step A fragments (16x32 bf16 each) for this wave's 16-row half.
// 16-bit A 16x32 layout: lanes 0-15 row=lane, K=[0..7],[16..23]; lanes 16-31 K=[8..15],[24..31].
__device__ __forceinline__ void load_a_frags(const unsigned short* aTile, int lane, int mt,
                                             Frag af[8])
{
    const int row = mt * 16 + (lane & 15);
    const unsigned short* base = aTile + row * LDSP + ((lane >> 4) << 3);
#pragma unroll
    for (int kk = 0; kk < 8; ++kk) {
        const unsigned short* p = base + kk * 32;
        af[kk].q[0] = *(const uint4*)(p);
        af[kk].q[1] = *(const uint4*)(p + 16);
    }
}

// C[32 x 256] += A * W^T using preloaded A fragments; W bf16 row-major [256 out][256 k].
__device__ __forceinline__ void gemm_frags_32x256(
    const Frag af[8], const unsigned short* __restrict__ W,
    int lane, int ntbase, v8f acc[4])
{
#pragma unroll
    for (int kk = 0; kk < 8; ++kk) {
        const int k0 = kk * 32;
#pragma unroll
        for (int j = 0; j < 4; ++j) {
            Frag ub;
            // 16-bit B 32x16 layout: col = lane&15; lanes 0-15 K=[k0..k0+7],[k0+16..k0+23]; lanes 16-31 K=[k0+8..k0+15],[k0+24..k0+31]
            const int col = (ntbase + j) * 16 + (lane & 15);
            const int kt  = k0 + ((lane >> 4) << 3);
            const unsigned short* p = W + col * H + kt;
            ub.q[0] = *(const uint4*)(p);
            ub.q[1] = *(const uint4*)(p + 16);
            acc[j] = __builtin_amdgcn_wmma_f32_16x16x32_bf16(
                false, af[kk].v, false, ub.v, (short)0, acc[j], false, false);
        }
    }
}

// ---------------- K0: convert five weight matrices f32 -> bf16 -------------
__global__ void k_cvt_w(const float* __restrict__ Wq, const float* __restrict__ Wk,
                        const float* __restrict__ Wv, const float* __restrict__ Wr,
                        const float* __restrict__ Wo, unsigned short* __restrict__ out)
{
    int i = blockIdx.x * blockDim.x + threadIdx.x;
    if (i >= H * H) return;
    out[0 * H * H + i] = f2bf(Wq[i]);
    out[1 * H * H + i] = f2bf(Wk[i]);
    out[2 * H * H + i] = f2bf(Wv[i]);
    out[3 * H * H + i] = f2bf(Wv[i] * 0.f + Wr[i]);
    out[4 * H * H + i] = f2bf(Wo[i]);
}

// ---------------- K1: q,k,v = x @ W^T (three GEMMs, shared A fragments) ----
__global__ __launch_bounds__(256) void k_qkv(
    const float* __restrict__ x,
    const unsigned short* __restrict__ Wqb, const unsigned short* __restrict__ Wkb,
    const unsigned short* __restrict__ Wvb,
    float* __restrict__ q, float* __restrict__ k, float* __restrict__ v)
{
    __shared__ __align__(16) unsigned short aTile[RPB * LDSP];
    const int tid = threadIdx.x, lane = tid & 31, wave = tid >> 5;
    const int row0 = blockIdx.x * RPB;
    for (int i = tid; i < RPB * H; i += 256) {
        int r = i >> 8, c = i & (H - 1);
        aTile[r * LDSP + c] = f2bf(x[(size_t)(row0 + r) * H + c]);
    }
    __syncthreads();
    const int mt = wave & 1, ntbase = (wave >> 1) * 4;
    Frag af[8];
    load_a_frags(aTile, lane, mt, af);           // loaded once, reused for 3 GEMMs
    const unsigned short* Ws[3] = { Wqb, Wkb, Wvb };
    float* Os[3] = { q, k, v };
    v8f zero = {};
#pragma unroll
    for (int widx = 0; widx < 3; ++widx) {
        v8f acc[4] = { zero, zero, zero, zero };
        gemm_frags_32x256(af, Ws[widx], lane, ntbase, acc);
#pragma unroll
        for (int j = 0; j < 4; ++j) {
            const int col = (ntbase + j) * 16 + (lane & 15);
#pragma unroll
            for (int r = 0; r < 8; ++r) {
                const int row = row0 + mt * 16 + r + ((lane >> 4) << 3);
                Os[widx][(size_t)row * H + col] = acc[j][r];
            }
        }
    }
}

// ---------------- K2: gq[m] = sum_n softmax_h(q * w_alpha*SCALE) * q -------
__global__ __launch_bounds__(256) void k_gq(
    const float* __restrict__ q, const float* __restrict__ w_alpha,
    float* __restrict__ gq)
{
    __shared__ float red[8];
    const int m = blockIdx.x, h = threadIdx.x;
    const int lane = h & 31, wave = h >> 5;
    const float was = w_alpha[h] * SCALE_F;
    float acc = 0.f;
    for (int n = 0; n < NB; ++n) {
        float qv = q[((size_t)m * NB + n) * H + h];
        float z = qv * was;
        float mx = z;
#pragma unroll
        for (int o = 16; o; o >>= 1) mx = fmaxf(mx, __shfl_xor(mx, o, 32));
        if (lane == 0) red[wave] = mx;
        __syncthreads();
        mx = red[0];
#pragma unroll
        for (int w2 = 1; w2 < 8; ++w2) mx = fmaxf(mx, red[w2]);
        __syncthreads();
        float e = __expf(z - mx);
        float s = e;
#pragma unroll
        for (int o = 16; o; o >>= 1) s += __shfl_xor(s, o, 32);
        if (lane == 0) red[wave] = s;
        __syncthreads();
        s = 0.f;
#pragma unroll
        for (int w2 = 0; w2 < 8; ++w2) s += red[w2];
        __syncthreads();
        acc += (e / s) * qv;
    }
    gq[(size_t)m * H + h] = acc;
}

// ------- K3: corr[m] = sum_{w<3} decay^w * ((gq[m] @ k[m-w]) @ v[m-w]) -----
__global__ __launch_bounds__(256) void k_corr(
    const float* __restrict__ k, const float* __restrict__ v,
    const float* __restrict__ gq, float* __restrict__ corr)
{
    __shared__ float gqs[H];
    __shared__ float t[H];
    const int m = blockIdx.x, tid = threadIdx.x;
    gqs[tid] = gq[(size_t)m * H + tid];
    float acc = 0.f, dsc = 1.f;
    for (int w = 0; w < CORR_WIN; ++w) {
        const int j = m - w;
        if (j < 0) break;                 // uniform per block
        __syncthreads();
        const float* kb = k + (size_t)j * NB * H;
        float tc = 0.f;
        for (int n = 0; n < NB; ++n) tc += gqs[n] * kb[(size_t)n * H + tid];
        t[tid] = tc;
        __syncthreads();
        const float* vb = v + (size_t)j * NB * H;
        float pd = 0.f;
        for (int c = 0; c < NB; ++c) pd += t[c] * vb[(size_t)c * H + tid];
        acc += dsc * pd;
        dsc *= DECAY_F;
    }
    corr[(size_t)m * H + tid] = acc;
}

// ---- K4: gk = wb*SCALE*gq^2*sum(k^2) + corr*gamma^(m+1)*gq*sum(k) ---------
__global__ __launch_bounds__(256) void k_gk(
    const float* __restrict__ k, const float* __restrict__ gq,
    const float* __restrict__ corr, const float* __restrict__ w_beta,
    float* __restrict__ gk)
{
    const int m = blockIdx.x, h = threadIdx.x;
    const float* kb = k + (size_t)m * NB * H;
    float s1 = 0.f, s2 = 0.f;
    for (int n = 0; n < NB; ++n) {
        float kv = kb[(size_t)n * H + h];
        s1 += kv; s2 += kv * kv;
    }
    const float g = gq[(size_t)m * H + h];
    const float gseq = __expf((float)(m + 1) * LNG_F);    // gamma^(m+1)
    gk[(size_t)m * H + h] = w_beta[h] * SCALE_F * g * g * s2
                          + corr[(size_t)m * H + h] * gseq * g * s1;
}

// ---- K5: ret = relu((gk ⊙ v) @ W_r^T + q), stored bf16 (reuses k buffer) --
__global__ __launch_bounds__(256) void k_ret(
    const float* __restrict__ v, const float* __restrict__ gk,
    const unsigned short* __restrict__ Wrb, const float* __restrict__ q,
    unsigned short* __restrict__ ret)
{
    __shared__ __align__(16) unsigned short aTile[RPB * LDSP];
    __shared__ float gks[H];
    const int tid = threadIdx.x, lane = tid & 31, wave = tid >> 5;
    const int m = blockIdx.x >> 3, rg = blockIdx.x & 7;
    const int row0 = m * NB + rg * RPB;
    gks[tid] = gk[(size_t)m * H + tid];
    __syncthreads();
    for (int i = tid; i < RPB * H; i += 256) {
        int r = i >> 8, c = i & (H - 1);
        aTile[r * LDSP + c] = f2bf(v[(size_t)(row0 + r) * H + c] * gks[c]);
    }
    __syncthreads();
    const int mt = wave & 1, ntbase = (wave >> 1) * 4;
    Frag af[8];
    load_a_frags(aTile, lane, mt, af);
    v8f zero = {};
    v8f acc[4] = { zero, zero, zero, zero };
    gemm_frags_32x256(af, Wrb, lane, ntbase, acc);
#pragma unroll
    for (int j = 0; j < 4; ++j) {
        const int col = (ntbase + j) * 16 + (lane & 15);
#pragma unroll
        for (int r = 0; r < 8; ++r) {
            const int row = row0 + mt * 16 + r + ((lane >> 4) << 3);
            float xv = acc[j][r] + q[(size_t)row * H + col];
            ret[(size_t)row * H + col] = f2bf(fmaxf(xv, 0.f));
        }
    }
}

// ---- K6: out = ret @ W_o^T + b_o, fused GroupNorm (1 block == 1 group) ----
// A tile is staged via CDNA5 async global->LDS DMA (ASYNCcnt-tracked).
__global__ __launch_bounds__(256) void k_out(
    const unsigned short* __restrict__ ret, const unsigned short* __restrict__ Wob,
    const float* __restrict__ b_o, const float* __restrict__ gn_w,
    const float* __restrict__ gn_b, float* __restrict__ out)
{
    __shared__ __align__(16) unsigned short aTile[RPB * LDSP];
    __shared__ float redS[8], redQ[8];
    const int tid = threadIdx.x, lane = tid & 31, wave = tid >> 5;
    const int m = blockIdx.x >> 3, grp = blockIdx.x & 7;
    const int row0 = m * NB + grp * RPB;

    // Async stage: 32x256 bf16 tile = 1024 x 16B chunks, 4 per thread.
#pragma unroll
    for (int i = 0; i < 4; ++i) {
        const int c   = tid + i * 256;          // chunk id
        const int r   = c >> 5;                 // row in tile
        const int cc  = (c & 31) << 3;          // column (bf16 elems)
        const unsigned short* src = ret + (size_t)(row0 + r) * H + cc;
        lds_us* dst = (lds_us*)(aTile + r * LDSP + cc);
        asm volatile("global_load_async_to_lds_b128 %0, %1, off"
                     :: "v"(dst), "v"(src) : "memory");
    }
    asm volatile("s_wait_asynccnt 0x0" ::: "memory");
    __syncthreads();

    const int mt = wave & 1, ntbase = (wave >> 1) * 4;
    Frag af[8];
    load_a_frags(aTile, lane, mt, af);
    v8f zero = {};
    v8f acc[4] = { zero, zero, zero, zero };
    gemm_frags_32x256(af, Wob, lane, ntbase, acc);

    float vals[32];
    float s = 0.f, sq = 0.f;
    int vi = 0;
#pragma unroll
    for (int j = 0; j < 4; ++j) {
        const int col = (ntbase + j) * 16 + (lane & 15);
        const float bo = b_o[col];
#pragma unroll
        for (int r = 0; r < 8; ++r) {
            float xv = acc[j][r] + bo;
            vals[vi++] = xv;
            s += xv; sq += xv * xv;
        }
    }
#pragma unroll
    for (int o = 16; o; o >>= 1) { s += __shfl_xor(s, o, 32); sq += __shfl_xor(sq, o, 32); }
    if (lane == 0) { redS[wave] = s; redQ[wave] = sq; }
    __syncthreads();
    float S = 0.f, Q = 0.f;
#pragma unroll
    for (int w2 = 0; w2 < 8; ++w2) { S += redS[w2]; Q += redQ[w2]; }
    const float inv = 1.0f / (float)(RPB * H);
    const float mu = S * inv;
    const float var = Q * inv - mu * mu;
    const float rsig = rsqrtf(var + EPS_F);

    vi = 0;
#pragma unroll
    for (int j = 0; j < 4; ++j) {
        const int col = (ntbase + j) * 16 + (lane & 15);
#pragma unroll
        for (int r = 0; r < 8; ++r) {
            const int rloc = mt * 16 + r + ((lane >> 4) << 3);
            const int nglob = grp * RPB + rloc;   // channel index n
            float y = (vals[vi++] - mu) * rsig * gn_w[nglob] + gn_b[nglob];
            out[(size_t)(1 + row0 + rloc) * H + col] = y;
        }
    }
    if (blockIdx.x == 0) out[tid] = 0.f;          // cached zero row
}

extern "C" void kernel_launch(void* const* d_in, const int* in_sizes, int n_in,
                              void* d_out, int out_size, void* d_ws, size_t ws_size,
                              hipStream_t stream) {
    (void)in_sizes; (void)n_in; (void)out_size; (void)ws_size;
    const float* message = (const float*)d_in[0];
    const float* W_q     = (const float*)d_in[1];
    const float* W_k     = (const float*)d_in[2];
    const float* W_v     = (const float*)d_in[3];
    const float* W_r     = (const float*)d_in[4];
    const float* W_o     = (const float*)d_in[5];
    const float* b_o     = (const float*)d_in[6];
    const float* w_alpha = (const float*)d_in[7];
    const float* w_beta  = (const float*)d_in[8];
    const float* gn_w    = (const float*)d_in[9];
    const float* gn_b    = (const float*)d_in[10];
    // d_in[11] = b_scope (contiguous uniform segments; implied by layout)

    float* q    = (float*)d_ws;
    float* k    = q + (size_t)TOTROWS * H;
    float* v    = k + (size_t)TOTROWS * H;
    float* gq   = v + (size_t)TOTROWS * H;
    float* corr = gq + (size_t)MMOL * H;
    float* gk   = corr + (size_t)MMOL * H;
    unsigned short* Wb  = (unsigned short*)(gk + (size_t)MMOL * H); // 5 * H*H bf16
    unsigned short* ret = (unsigned short*)k;                       // reuse k after k_gk

    k_cvt_w<<<(H * H + 255) / 256, 256, 0, stream>>>(W_q, W_k, W_v, W_r, W_o, Wb);
    k_qkv<<<TOTROWS / RPB, 256, 0, stream>>>(message, Wb, Wb + H * H, Wb + 2 * H * H, q, k, v);
    k_gq<<<MMOL, 256, 0, stream>>>(q, w_alpha, gq);
    k_corr<<<MMOL, 256, 0, stream>>>(k, v, gq, corr);
    k_gk<<<MMOL, 256, 0, stream>>>(k, gq, corr, w_beta, gk);
    k_ret<<<MMOL * 8, 256, 0, stream>>>(v, gk, Wb + 3 * H * H, q, ret);
    k_out<<<MMOL * 8, 256, 0, stream>>>(ret, Wb + 4 * H * H, b_o, gn_w, gn_b, (float*)d_out);
}
